// NystromTransformerLayer_49091476193799
// MI455X (gfx1250) — compile-verified, matrix operand order checked by reading, not run on hardware
//
#include <hip/hip_runtime.h>
#include <hip/hip_bf16.h>
#include <cstdint>

// ---------------------------------------------------------------------------
// CDNA5 (gfx1250) Nystrom attention layer. wave32, bf16 WMMA w/ f32 accum,
// async global->LDS staging; all WMMA operands stored k-contiguous so every
// fragment load is 2x b128.
// ---------------------------------------------------------------------------

typedef __bf16 bf16_t;
typedef __attribute__((ext_vector_type(16))) __bf16 v16bf;
typedef __attribute__((ext_vector_type(8)))  float  v8f;
typedef __attribute__((ext_vector_type(4)))  int    v4i;

#define DEV __device__ __forceinline__

#ifndef __has_builtin
#define __has_builtin(x) 0
#endif

#if __has_builtin(__builtin_amdgcn_global_load_async_to_lds_b128)
#define HAVE_ASYNC 1
#else
#define HAVE_ASYNC 0
#endif

typedef __attribute__((address_space(1))) v4i* gv4i_p;   // global int4*
typedef __attribute__((address_space(3))) v4i* lv4i_p;   // LDS int4*

DEV int lane_id() { return (int)(threadIdx.x & 31u); }
DEV int wave_id() { return (int)(threadIdx.x >> 5); }

DEV v8f vzero8() { v8f z = {0.f,0.f,0.f,0.f,0.f,0.f,0.f,0.f}; return z; }

DEV v8f wmma_bf16(v16bf a, v16bf b, v8f c) {
  return __builtin_amdgcn_wmma_f32_16x16x32_bf16(false, a, false, b, (short)0, c,
                                                 false, false);
}

// 16B global -> LDS copy: async (ASYNCcnt) if available, sync fallback
DEV void copy16_g2l(const void* g, void* l) {
#if HAVE_ASYNC
  __builtin_amdgcn_global_load_async_to_lds_b128((gv4i_p)g, (lv4i_p)l, 0, 0);
#else
  *(int4*)l = *(const int4*)g;
#endif
}

template <int K> DEV void wait_async() {
#if HAVE_ASYNC
# if __has_builtin(__builtin_amdgcn_s_wait_asynccnt)
  __builtin_amdgcn_s_wait_asynccnt(K);
# else
  asm volatile("s_wait_asynccnt %0" :: "i"(K) : "memory");
# endif
#endif
}

// ---- fragment loaders (ISA 7.12.2 layouts, wave32) -------------------------
// A-style fragment (16x32 bf16, k-contiguous rows): lane l -> row = l&15;
// slot j -> k = (l>>4)*8 + (j&7) + (j>=8?16:0). Used for:
//   * A operands from row-major [M][K]
//   * B operands of A@B^T from row-major [N][K]
//   * B operands of A@B from COLUMN-major [N][K] storage
DEV v16bf frag_a_bf(const bf16_t* __restrict__ A, int lda) {
  const int l = lane_id(), r = l & 15, h = l >> 4;
  const bf16_t* p = A + (size_t)r * lda + h * 8;
  v16bf a;
#pragma unroll
  for (int j = 0; j < 8; ++j) a[j] = p[j];
#pragma unroll
  for (int j = 0; j < 8; ++j) a[j + 8] = p[16 + j];
  return a;
}

// D: VGPR i -> row = i + 8*(l>>4), col = l&15
DEV void store_d_f32(float* __restrict__ C, int ldc, v8f d, float scale) {
  const int l = lane_id(), n = l & 15, h = l >> 4;
#pragma unroll
  for (int i = 0; i < 8; ++i) C[(size_t)(i + 8 * h) * ldc + n] = d[i] * scale;
}

// ===========================================================================
// K0a: f32 -> bf16 elementwise convert
// ===========================================================================
__global__ void k_cvt(const float* __restrict__ s, bf16_t* __restrict__ d, int n) {
  int tid = blockIdx.x * 256 + threadIdx.x;
  if (tid < n) d[tid] = (bf16_t)s[tid];
}

// K0b: f32 [I=512][O] -> bf16 transposed [O][512] (weights: one-time)
__global__ void k_cvt_t(const float* __restrict__ s, bf16_t* __restrict__ d, int O) {
  int tid = blockIdx.x * 256 + threadIdx.x;     // over 512*O
  int i = tid & 511, o = tid >> 9;
  d[(size_t)o * 512 + i] = (bf16_t)s[(size_t)i * O + o];
}

// ===========================================================================
// K1: qkv = x @ W_qkv. Block: 128 rows x 64-col strip. B tile (32k x 64 cols)
// staged column-major in LDS (pad 40) via async b128, double buffered.
// ===========================================================================
__global__ __launch_bounds__(256) void k_qkv(
    const bf16_t* __restrict__ xbf, const bf16_t* __restrict__ Wt,  // Wt:[1536][512]
    bf16_t* __restrict__ qbf, bf16_t* __restrict__ kbf,
    bf16_t* __restrict__ vbfT, float* __restrict__ vf) {
  __shared__ __align__(16) bf16_t Bs[2][64 * 40];
  const int mb = blockIdx.x / 24, ts = blockIdx.x % 24;   // uniform strip
  const int w = wave_id(), tm = mb * 8 + w;
  const bf16_t* Ab = xbf + (size_t)tm * 16 * 512;
  const int t = threadIdx.x, c = t >> 2, ck = (t & 3) * 8;

  auto issue = [&](int buf, int kk) {
    copy16_g2l(Wt + (size_t)(ts * 64 + c) * 512 + kk + ck, &Bs[buf][c * 40 + ck]);
  };
  issue(0, 0);
  v8f acc[4] = {vzero8(), vzero8(), vzero8(), vzero8()};
  int cur = 0;
  for (int kk = 0; kk < 512; kk += 32) {
    bool more = (kk + 32) < 512;
    if (more) issue(cur ^ 1, kk + 32);
    if (more) wait_async<1>(); else wait_async<0>();
    __syncthreads();
    v16bf a = frag_a_bf(Ab + kk, 512);
#pragma unroll
    for (int dn = 0; dn < 4; ++dn) {
      v16bf b = frag_a_bf(&Bs[cur][(dn * 16) * 40], 40);
      acc[dn] = wmma_bf16(a, b, acc[dn]);
    }
    __syncthreads();
    cur ^= 1;
  }
  const int l = lane_id(), n = l & 15, h = l >> 4;
  const int tq = (ts * 64) >> 9;                          // 0:q 1:k 2:v (uniform)
#pragma unroll
  for (int dn = 0; dn < 4; ++dn) {
    int C = ts * 64 + dn * 16 + n;
    int rem = C & 511, hh = rem >> 6, dd = rem & 63;
    int R0 = tm * 16 + 8 * h;                             // first of 8 rows
    int bb = R0 >> 12, nn0 = R0 & 4095;
    size_t ch = (size_t)(bb * 8 + hh);
    if (tq == 0) {
#pragma unroll
      for (int i = 0; i < 8; ++i)
        qbf[(ch * 4096 + nn0 + i) * 64 + dd] = (bf16_t)(acc[dn][i] * 0.125f);
    } else if (tq == 1) {
#pragma unroll
      for (int i = 0; i < 8; ++i)
        kbf[(ch * 4096 + nn0 + i) * 64 + dd] = (bf16_t)acc[dn][i];
    } else {
      alignas(16) bf16_t t8[8];
#pragma unroll
      for (int i = 0; i < 8; ++i) {
        vf[(ch * 4096 + nn0 + i) * 64 + dd] = acc[dn][i];
        t8[i] = (bf16_t)acc[dn][i];
      }
      *(int4*)(vbfT + (ch * 64 + dd) * 4096 + nn0) = *(const int4*)t8;  // v^T
    }
  }
}

// ===========================================================================
// K2: landmark means (l = 16 tokens / landmark)
// ===========================================================================
__global__ void k_landmarks(const bf16_t* __restrict__ qbf,
                            const bf16_t* __restrict__ kbf,
                            bf16_t* __restrict__ qland, bf16_t* __restrict__ kland) {
  size_t tid = (size_t)blockIdx.x * 256 + threadIdx.x;    // 2*32*256*64
  int which = tid >= (size_t)32 * 256 * 64;
  size_t u = tid & ((size_t)32 * 256 * 64 - 1);
  int d = (int)(u & 63); size_t r = u >> 6;
  int m = (int)(r & 255); int bh = (int)(r >> 8);
  const bf16_t* p = (which ? kbf : qbf) + (((size_t)bh * 4096) + (size_t)m * 16) * 64 + d;
  float s = 0.f;
#pragma unroll
  for (int j = 0; j < 16; ++j) s += (float)p[(size_t)j * 64];
  (which ? kland : qland)[((size_t)bh * 256 + m) * 64 + d] = (bf16_t)(s * 0.0625f);
}

// ===========================================================================
// K3: sim2 = q_land @ k_land^T  (per bh, 256x256, K=64)
// ===========================================================================
__global__ __launch_bounds__(256) void k_sim2(
    const bf16_t* __restrict__ qland, const bf16_t* __restrict__ kland,
    float* __restrict__ sim2) {
  int wid = blockIdx.x * 8 + wave_id();
  int bh = wid >> 8, tt = wid & 255, tm = tt >> 4, tn = tt & 15;
  const bf16_t* Ab = qland + (size_t)bh * 256 * 64 + (size_t)tm * 16 * 64;
  const bf16_t* Bb = kland + (size_t)bh * 256 * 64 + (size_t)tn * 16 * 64;
  v8f acc = vzero8();
  acc = wmma_bf16(frag_a_bf(Ab, 64),      frag_a_bf(Bb, 64),      acc);
  acc = wmma_bf16(frag_a_bf(Ab + 32, 64), frag_a_bf(Bb + 32, 64), acc);
  store_d_f32(sim2 + (size_t)bh * 65536 + (size_t)tm * 16 * 256 + tn * 16, 256, acc, 1.f);
}

// ===========================================================================
// K4: row softmax over 256 -> f32 + bf16
// ===========================================================================
__global__ __launch_bounds__(256) void k_softmax256(
    const float* __restrict__ S, float* __restrict__ P, bf16_t* __restrict__ Pb) {
  size_t row = (size_t)blockIdx.x * 8 + wave_id();
  const float* p = S + row * 256;
  int l = lane_id();
  float v[8];
#pragma unroll
  for (int j = 0; j < 8; ++j) v[j] = p[l * 8 + j];
  float mx = v[0];
#pragma unroll
  for (int j = 1; j < 8; ++j) mx = fmaxf(mx, v[j]);
  for (int m = 1; m < 32; m <<= 1) mx = fmaxf(mx, __shfl_xor(mx, m, 32));
  float s = 0.f;
#pragma unroll
  for (int j = 0; j < 8; ++j) { v[j] = __expf(v[j] - mx); s += v[j]; }
  for (int m = 1; m < 32; m <<= 1) s += __shfl_xor(s, m, 32);
  float inv = 1.f / s;
#pragma unroll
  for (int j = 0; j < 8; ++j) {
    float o = v[j] * inv;
    P[row * 256 + l * 8 + j] = o;
    Pb[row * 256 + l * 8 + j] = (bf16_t)o;
  }
}

// ===========================================================================
// K5..K7: pinv init (global max row/col sums, z = x^T/(c1*c2))
// ===========================================================================
__global__ void k_init_scal(unsigned* g) { if (threadIdx.x < 2) g[threadIdx.x] = 0u; }

__global__ __launch_bounds__(256) void k_maxsums(const float* __restrict__ attn2,
                                                 unsigned* __restrict__ g) {
  int idx = blockIdx.x * 8 + wave_id();
  int bh = idx >> 8, r = idx & 255;
  const float* base = attn2 + (size_t)bh * 65536;
  int l = lane_id();
  float s1 = 0.f, s2 = 0.f;
#pragma unroll
  for (int j = 0; j < 8; ++j) s1 += fabsf(base[(size_t)r * 256 + l * 8 + j]);
#pragma unroll
  for (int j = 0; j < 8; ++j) s2 += fabsf(base[(size_t)(l * 8 + j) * 256 + r]);
  for (int m = 1; m < 32; m <<= 1) { s1 += __shfl_xor(s1, m, 32); s2 += __shfl_xor(s2, m, 32); }
  if (l == 0) {
    atomicMax(g + 0, __float_as_uint(s1));
    atomicMax(g + 1, __float_as_uint(s2));
  }
}

// z row-major AND column-major. Note z^T(colmajor) == attn2/denom elementwise.
__global__ void k_pinv_init(const float* __restrict__ attn2,
                            const unsigned* __restrict__ g,
                            bf16_t* __restrict__ z, bf16_t* __restrict__ zT) {
  size_t tid = (size_t)blockIdx.x * 256 + threadIdx.x;    // (bh, a, b)
  int b = (int)(tid & 255); size_t r = tid >> 8;
  int a = (int)(r & 255); int bh = (int)(r >> 8);
  float denom = __uint_as_float(g[0]) * __uint_as_float(g[1]);
  float v = attn2[tid] / denom;                           // = z[b][a]
  zT[tid] = (bf16_t)v;                                    // colmajor: [a][b] slot
  z[((size_t)bh * 256 + b) * 256 + a] = (bf16_t)v;        // rowmajor
}

// ===========================================================================
// K8: per-bh GEMM, P = A@B; out = outscale*(sX*X - P) (X null -> outscale*P).
// A rowmajor [256][256]; B COLUMN-major [N][256]; outputs: Crm rowmajor
// and/or Ccm colmajor (16B packed stores). B staged colmajor in LDS (pad 40).
// ===========================================================================
template <int N>
__global__ __launch_bounds__(256) void k_gemm256(
    const bf16_t* __restrict__ A, const bf16_t* __restrict__ Bcm,
    const bf16_t* __restrict__ X, bf16_t* __restrict__ Crm,
    bf16_t* __restrict__ Ccm, float sX, float outscale) {
  constexpr int S = N / 64;          // strips per tm row-block
  constexpr int TMB = 8 / S;         // tm values per block
  constexpr int CNT = N / 64;        // async b128s per thread per buffer
  __shared__ __align__(16) bf16_t Bs[2][N * 40];
  const int bh = blockIdx.x / (16 / TMB);
  const int tmb = blockIdx.x % (16 / TMB);
  const int w = wave_id();
  const int tm = tmb * TMB + w / S;
  const int strip = (w % S) * 64;
  const bf16_t* Ab = A + (size_t)bh * 65536 + (size_t)tm * 16 * 256;
  const bf16_t* Bb = Bcm + (size_t)bh * 256 * N;
  const int t = threadIdx.x;

  auto issue = [&](int buf, int kk) {
#pragma unroll
    for (int c4 = 0; c4 < CNT; ++c4) {
      int u = c4 * 256 + t;
      int c = u >> 2, ck = (u & 3) * 8;
      copy16_g2l(Bb + (size_t)c * 256 + kk + ck, &Bs[buf][c * 40 + ck]);
    }
  };
  issue(0, 0);
  v8f acc[4] = {vzero8(), vzero8(), vzero8(), vzero8()};
  int cur = 0;
  for (int kk = 0; kk < 256; kk += 32) {
    bool more = (kk + 32) < 256;
    if (more) issue(cur ^ 1, kk + 32);
    if (more) wait_async<CNT>(); else wait_async<0>();
    __syncthreads();
    v16bf a = frag_a_bf(Ab + kk, 256);
#pragma unroll
    for (int dn = 0; dn < 4; ++dn) {
      v16bf b = frag_a_bf(&Bs[cur][(strip + dn * 16) * 40], 40);
      acc[dn] = wmma_bf16(a, b, acc[dn]);
    }
    __syncthreads();
    cur ^= 1;
  }
  const int l = lane_id(), n = l & 15, h = l >> 4;
#pragma unroll
  for (int dn = 0; dn < 4; ++dn) {
    int col = strip + dn * 16 + n;
    float vv[8];
#pragma unroll
    for (int i = 0; i < 8; ++i) {
      size_t rm = (size_t)bh * 256 * N + (size_t)(tm * 16 + i + 8 * h) * N + col;
      float v = acc[dn][i];
      if (X) v = sX * (float)X[rm] - v;
      vv[i] = v * outscale;
      if (Crm) Crm[rm] = (bf16_t)vv[i];
    }
    if (Ccm) {
      alignas(16) bf16_t t8[8];
#pragma unroll
      for (int i = 0; i < 8; ++i) t8[i] = (bf16_t)vv[i];
      *(int4*)(Ccm + (size_t)bh * 256 * N + (size_t)col * 256 + tm * 16 + 8 * h) =
          *(const int4*)t8;
    }
  }
}

// ===========================================================================
// K9: row stats (max, sumexp) of S = A[Mx64] @ B[Lx64]^T per bh
// ===========================================================================
__global__ __launch_bounds__(256) void k_rowstats(
    const bf16_t* __restrict__ A, const bf16_t* __restrict__ Bt,
    float* __restrict__ rowmax, float* __restrict__ rowsum, int M, int L) {
  int mt = M >> 4;
  int wid = blockIdx.x * 8 + wave_id();
  int bh = wid / mt, tm = wid % mt;
  const bf16_t* Ab = A + (size_t)bh * M * 64 + (size_t)tm * 16 * 64;
  const bf16_t* Bb = Bt + (size_t)bh * L * 64;
  v16bf a0 = frag_a_bf(Ab, 64), a1 = frag_a_bf(Ab + 32, 64);
  float rmax[8];
#pragma unroll
  for (int i = 0; i < 8; ++i) rmax[i] = -3.4e38f;
  for (int tn = 0; tn < (L >> 4); ++tn) {
    v8f s = vzero8();
    s = wmma_bf16(a0, frag_a_bf(Bb + (size_t)tn * 16 * 64, 64), s);
    s = wmma_bf16(a1, frag_a_bf(Bb + (size_t)tn * 16 * 64 + 32, 64), s);
#pragma unroll
    for (int i = 0; i < 8; ++i) rmax[i] = fmaxf(rmax[i], s[i]);
  }
  for (int m = 1; m < 16; m <<= 1)
#pragma unroll
    for (int i = 0; i < 8; ++i) rmax[i] = fmaxf(rmax[i], __shfl_xor(rmax[i], m, 32));
  float rsum[8];
#pragma unroll
  for (int i = 0; i < 8; ++i) rsum[i] = 0.f;
  for (int tn = 0; tn < (L >> 4); ++tn) {
    v8f s = vzero8();
    s = wmma_bf16(a0, frag_a_bf(Bb + (size_t)tn * 16 * 64, 64), s);
    s = wmma_bf16(a1, frag_a_bf(Bb + (size_t)tn * 16 * 64 + 32, 64), s);
#pragma unroll
    for (int i = 0; i < 8; ++i) rsum[i] += __expf(s[i] - rmax[i]);
  }
  for (int m = 1; m < 16; m <<= 1)
#pragma unroll
    for (int i = 0; i < 8; ++i) rsum[i] += __shfl_xor(rsum[i], m, 32);
  const int l = lane_id(), h = l >> 4;
  if ((l & 15) == 0) {
#pragma unroll
    for (int i = 0; i < 8; ++i) {
      int row = tm * 16 + i + 8 * h;
      rowmax[(size_t)bh * M + row] = rmax[i];
      rowsum[(size_t)bh * M + row] = rsum[i];
    }
  }
}

// ===========================================================================
// K10: Out = diag(1/rowsum) * exp(A@B^T - rowmax) @ V, L streamed by 32.
// Vt is V TRANSPOSED: [bh][64][L] (fragments k-contiguous, no LDS needed).
// Outf rowmajor [bh][M][64]; Outb colmajor [bh][64][M] (packed 16B stores).
// ===========================================================================
__global__ __launch_bounds__(256) void k_softmax_av(
    const bf16_t* __restrict__ A, const bf16_t* __restrict__ Bt,
    const bf16_t* __restrict__ Vt, const float* __restrict__ rowmax,
    const float* __restrict__ rowsum, float* __restrict__ Outf,
    bf16_t* __restrict__ Outb, int M, int L) {
  __shared__ __align__(16) bf16_t lds[8][16][40];   // 80B row pitch (16B aligned)
  int w = wave_id();
  int mt = M >> 4;
  int wid = blockIdx.x * 8 + w;
  int bh = wid / mt, tm = wid % mt;
  const int l = lane_id(), n = l & 15, h = l >> 4;
  const bf16_t* Ab = A + (size_t)bh * M * 64 + (size_t)tm * 16 * 64;
  const bf16_t* Bb = Bt + (size_t)bh * L * 64;
  const bf16_t* Vb = Vt + (size_t)bh * 64 * L;
  v16bf a0 = frag_a_bf(Ab, 64), a1 = frag_a_bf(Ab + 32, 64);
  float rm[8], rs[8];
#pragma unroll
  for (int i = 0; i < 8; ++i) {
    int row = tm * 16 + i + 8 * h;
    rm[i] = rowmax[(size_t)bh * M + row];
    rs[i] = 1.f / rowsum[(size_t)bh * M + row];
  }
  v8f acc[4] = {vzero8(), vzero8(), vzero8(), vzero8()};
  for (int jc = 0; jc < (L >> 5); ++jc) {
#pragma unroll
    for (int sub = 0; sub < 2; ++sub) {
      int tn = jc * 2 + sub;
      v8f s = vzero8();
      s = wmma_bf16(a0, frag_a_bf(Bb + (size_t)tn * 16 * 64, 64), s);
      s = wmma_bf16(a1, frag_a_bf(Bb + (size_t)tn * 16 * 64 + 32, 64), s);
#pragma unroll
      for (int i = 0; i < 8; ++i)
        lds[w][i + 8 * h][sub * 16 + n] = (bf16_t)__expf(s[i] - rm[i]);
    }
    v16bf ap = frag_a_bf(&lds[w][0][0], 40);   // same-wave DS RAW: in-order
#pragma unroll
    for (int dn = 0; dn < 4; ++dn) {
      v16bf bv = frag_a_bf(Vb + (size_t)(dn * 16) * L + jc * 32, L);
      acc[dn] = wmma_bf16(ap, bv, acc[dn]);
    }
  }
#pragma unroll
  for (int dn = 0; dn < 4; ++dn) {
    if (Outf) {
#pragma unroll
      for (int i = 0; i < 8; ++i)
        Outf[(size_t)bh * M * 64 + (size_t)(tm * 16 + i + 8 * h) * 64 + dn * 16 + n] =
            acc[dn][i] * rs[i];
    }
    if (Outb) {
      alignas(16) bf16_t t8[8];
#pragma unroll
      for (int i = 0; i < 8; ++i) t8[i] = (bf16_t)(acc[dn][i] * rs[i]);
      *(int4*)(Outb + ((size_t)bh * 64 + dn * 16 + n) * M + tm * 16 + 8 * h) =
          *(const int4*)t8;
    }
  }
}

// ===========================================================================
// K11: depthwise residual conv (k=33, same) + head merge -> bf16
// ===========================================================================
__global__ void k_conv_merge(const float* __restrict__ vf,
                             const float* __restrict__ out_head,
                             const float* __restrict__ Wconv,
                             bf16_t* __restrict__ Obf) {
  size_t tid = (size_t)blockIdx.x * 256 + threadIdx.x;    // 32*4096*64
  int d = (int)(tid & 63); size_t r = tid >> 6;
  int nn = (int)(r & 4095); int bh = (int)(r >> 12);
  int hh = bh & 7;
  float acc = out_head[tid];
  const float* vcol = vf + ((size_t)bh * 4096) * 64 + d;
#pragma unroll
  for (int t = 0; t < 33; ++t) {
    int ns = nn + t - 16;
    if (ns >= 0 && ns < 4096) acc += Wconv[hh * 33 + t] * vcol[(size_t)ns * 64];
  }
  int b = bh >> 3;
  Obf[(((size_t)b * 4096 + nn) * 512) + hh * 64 + d] = (bf16_t)acc;
}

// ===========================================================================
// K12: y = x + O @ W_out + b_out. Same LDS-staged structure as k_qkv.
// ===========================================================================
__global__ __launch_bounds__(256) void k_outproj(
    const bf16_t* __restrict__ Obf, const bf16_t* __restrict__ Wt, // Wt:[512][512]
    const float* __restrict__ bout, const float* __restrict__ x,
    float* __restrict__ y) {
  __shared__ __align__(16) bf16_t Bs[2][64 * 40];
  const int mb = blockIdx.x / 8, ts = blockIdx.x % 8;
  const int w = wave_id(), tm = mb * 8 + w;
  const bf16_t* Ab = Obf + (size_t)tm * 16 * 512;
  const int t = threadIdx.x, c = t >> 2, ck = (t & 3) * 8;

  auto issue = [&](int buf, int kk) {
    copy16_g2l(Wt + (size_t)(ts * 64 + c) * 512 + kk + ck, &Bs[buf][c * 40 + ck]);
  };
  issue(0, 0);
  v8f acc[4] = {vzero8(), vzero8(), vzero8(), vzero8()};
  int cur = 0;
  for (int kk = 0; kk < 512; kk += 32) {
    bool more = (kk + 32) < 512;
    if (more) issue(cur ^ 1, kk + 32);
    if (more) wait_async<1>(); else wait_async<0>();
    __syncthreads();
    v16bf a = frag_a_bf(Ab + kk, 512);
#pragma unroll
    for (int dn = 0; dn < 4; ++dn) {
      v16bf b = frag_a_bf(&Bs[cur][(dn * 16) * 40], 40);
      acc[dn] = wmma_bf16(a, b, acc[dn]);
    }
    __syncthreads();
    cur ^= 1;
  }
  const int l = lane_id(), n = l & 15, h = l >> 4;
#pragma unroll
  for (int dn = 0; dn < 4; ++dn) {
    int C = ts * 64 + dn * 16 + n;
    float bc = bout[C];
#pragma unroll
    for (int i = 0; i < 8; ++i) {
      size_t R = (size_t)tm * 16 + i + 8 * h;
      y[R * 512 + C] = acc[dn][i] + bc + x[R * 512 + C];
    }
  }
}

// ===========================================================================
extern "C" void kernel_launch(void* const* d_in, const int* in_sizes, int n_in,
                              void* d_out, int out_size, void* d_ws, size_t ws_size,
                              hipStream_t stream) {
  (void)in_sizes; (void)n_in; (void)out_size; (void)ws_size;
  const float* x     = (const float*)d_in[0];
  const float* Wqkv  = (const float*)d_in[1];
  const float* Wout  = (const float*)d_in[2];
  const float* bout  = (const float*)d_in[3];
  const float* Wconv = (const float*)d_in[4];
  float* y = (float*)d_out;

  char* ws = (char*)d_ws;
  size_t off = 0;
  auto alloc = [&](size_t bytes) -> void* {
    void* p = ws + off;
    off = (off + bytes + 255) & ~(size_t)255;
    return p;
  };

  const size_t BHND = (size_t)32 * 4096 * 64;
  const size_t BHMD = (size_t)32 * 256 * 64;
  const size_t BHMM = (size_t)32 * 256 * 256;

  bf16_t* xbf     = (bf16_t*)alloc((size_t)16384 * 512 * 2);
  bf16_t* WqkvT   = (bf16_t*)alloc((size_t)1536 * 512 * 2);
  bf16_t* WoutT   = (bf16_t*)alloc((size_t)512 * 512 * 2);
  bf16_t* qbf     = (bf16_t*)alloc(BHND * 2);
  bf16_t* kbf     = (bf16_t*)alloc(BHND * 2);
  bf16_t* vbfT    = (bf16_t*)alloc(BHND * 2);
  float*  vf      = (float*) alloc(BHND * 4);
  bf16_t* qland   = (bf16_t*)alloc(BHMD * 2);
  bf16_t* kland   = (bf16_t*)alloc(BHMD * 2);
  float*  sim2    = (float*) alloc(BHMM * 4);
  float*  attn2   = (float*) alloc(BHMM * 4);
  bf16_t* attn2bf = (bf16_t*)alloc(BHMM * 2);
  bf16_t* z1      = (bf16_t*)alloc(BHMM * 2);
  bf16_t* z2      = (bf16_t*)alloc(BHMM * 2);
  bf16_t* z1T     = (bf16_t*)alloc(BHMM * 2);
  bf16_t* z2T     = (bf16_t*)alloc(BHMM * 2);
  bf16_t* xzbf    = (bf16_t*)alloc(BHMM * 2);
  bf16_t* xzbfT   = (bf16_t*)alloc(BHMM * 2);
  bf16_t* t2T     = (bf16_t*)alloc(BHMM * 2);
  bf16_t* t4T     = (bf16_t*)alloc(BHMM * 2);
  unsigned* gmax  = (unsigned*)alloc(256);
  float*  rmax3   = (float*) alloc((size_t)32 * 256 * 4);
  float*  rsum3   = (float*) alloc((size_t)32 * 256 * 4);
  bf16_t* attn3vT = (bf16_t*)alloc(BHMD * 2);
  bf16_t* W2T     = (bf16_t*)alloc(BHMD * 2);
  float*  rmax1   = (float*) alloc((size_t)32 * 4096 * 4);
  float*  rsum1   = (float*) alloc((size_t)32 * 4096 * 4);
  float*  ohead   = (float*) alloc(BHND * 4);
  bf16_t* Obf     = (bf16_t*)alloc((size_t)16384 * 512 * 2);

  // 0. pre-convert: x -> bf16; weights -> bf16 TRANSPOSED (k-contiguous B)
  k_cvt<<<32768, 256, 0, stream>>>(x, xbf, 16384 * 512);
  k_cvt_t<<<3072, 256, 0, stream>>>(Wqkv, WqkvT, 1536);
  k_cvt_t<<<1024, 256, 0, stream>>>(Wout, WoutT, 512);
  // 1. QKV projection
  k_qkv<<<3072, 256, 0, stream>>>(xbf, WqkvT, qbf, kbf, vbfT, vf);
  // 2. landmark means
  k_landmarks<<<4096, 256, 0, stream>>>(qbf, kbf, qland, kland);
  // 3. sim2 + softmax
  k_sim2<<<1024, 256, 0, stream>>>(qland, kland, sim2);
  k_softmax256<<<1024, 256, 0, stream>>>(sim2, attn2, attn2bf);
  // 4. pinv init
  k_init_scal<<<1, 32, 0, stream>>>(gmax);
  k_maxsums<<<1024, 256, 0, stream>>>(attn2, gmax);
  k_pinv_init<<<8192, 256, 0, stream>>>(attn2, gmax, z1, z1T);
  // 5. attn3v^T = (softmax(q_land @ k^T) @ v)^T  (M=256, L=4096)
  k_rowstats<<<64, 256, 0, stream>>>(qland, kbf, rmax3, rsum3, 256, 4096);
  k_softmax_av<<<64, 256, 0, stream>>>(qland, kbf, vbfT, rmax3, rsum3,
                                       nullptr, attn3vT, 256, 4096);
  // 6. Newton-Schulz:
  //    xz = a@z; t2 = 7xz - xz@xz; t4 = 15xz - xz@t2; z' = 0.25(13z - z@t4)
  bf16_t* zin = z1;  bf16_t* zinT = z1T;
  bf16_t* zout = z2; bf16_t* zoutT = z2T;
  for (int it = 0; it < 6; ++it) {
    k_gemm256<256><<<256, 256, 0, stream>>>(attn2bf, zinT, nullptr, xzbf, xzbfT, 0.f, 1.f);
    k_gemm256<256><<<256, 256, 0, stream>>>(xzbf, xzbfT, xzbf, nullptr, t2T, 7.f,  1.f);
    k_gemm256<256><<<256, 256, 0, stream>>>(xzbf, t2T,   xzbf, nullptr, t4T, 15.f, 1.f);
    k_gemm256<256><<<256, 256, 0, stream>>>(zin,  t4T,   zin,  zout,    zoutT, 13.f, 0.25f);
    bf16_t* a = zin;  zin = zout;  zout = a;
    bf16_t* b = zinT; zinT = zoutT; zoutT = b;
  }
  // 7. W2^T = (z @ attn3v)^T
  k_gemm256<64><<<64, 256, 0, stream>>>(zin, attn3vT, nullptr, nullptr, W2T, 0.f, 1.f);
  // 8. out_head = softmax(q @ k_land^T) @ W2  (M=4096, L=256)
  k_rowstats<<<1024, 256, 0, stream>>>(qbf, kland, rmax1, rsum1, 4096, 256);
  k_softmax_av<<<1024, 256, 0, stream>>>(qbf, kland, W2T, rmax1, rsum1,
                                         ohead, nullptr, 4096, 256);
  // 9. depthwise conv residual + merge heads
  k_conv_merge<<<32768, 256, 0, stream>>>(vf, ohead, Wconv, Obf);
  // 10. y = x + O @ W_out + b_out
  k_outproj<<<1024, 256, 0, stream>>>(Obf, WoutT, bout, x, y);
}
